// Gating_37598143709808
// MI455X (gfx1250) — matrix operand order repaired and checked by reading, not compiled, not run on hardware
//
#include <hip/hip_runtime.h>
#include <hip/hip_bf16.h>
#include <math.h>

// Problem constants (match reference)
#define B_ROWS 262144
#define DDIM   256
#define HDIM   256
#define EDIM   5

typedef __attribute__((ext_vector_type(16))) __bf16       v16bf;
typedef __attribute__((ext_vector_type(8)))  __bf16       v8bf;
typedef __attribute__((ext_vector_type(8)))  float        v8f;
typedef __attribute__((ext_vector_type(4)))  unsigned int u32x4;
typedef __attribute__((ext_vector_type(4)))  int          v4i;

// LDS layout: W1^T as bf16, 256 rows x (512B data + 16B pad) then W2 f32 [256][5]
#define LDS_W1_STRIDE 528
#define LDS_W1_BYTES  (256 * LDS_W1_STRIDE)
#define LDS_W2_OFF    LDS_W1_BYTES
#define LDS_BYTES     (LDS_W1_BYTES + DDIM * EDIM * 4)

#if defined(__has_builtin)
#if __has_builtin(__builtin_amdgcn_global_load_async_to_lds_b128)
#define HAVE_ASYNC_LDS 1
#endif
#if __has_builtin(__builtin_amdgcn_s_wait_asynccnt)
#define HAVE_WAIT_ASYNC 1
#endif
#endif

// Stage W2 (5120 bytes) into LDS. Use the CDNA5 async global->LDS path when the
// builtin is available; otherwise plain load/store.
static __device__ inline void stage_w2(const float* __restrict__ W2, char* smem, int tid)
{
#if defined(HAVE_ASYNC_LDS)
  // 5120 B / 16 B = 320 b128 chunks
  for (int c = tid; c < (DDIM * EDIM * 4) / 16; c += 256) {
    __attribute__((address_space(1))) v4i* g =
        (__attribute__((address_space(1))) v4i*)(unsigned long long)(const void*)(W2 + c * 4);
    __attribute__((address_space(3))) v4i* l =
        (__attribute__((address_space(3))) v4i*)(unsigned int)(unsigned long long)(const void*)(smem + LDS_W2_OFF + c * 16);
    __builtin_amdgcn_global_load_async_to_lds_b128(g, l, 0, 0);
  }
#else
  float* sW2 = (float*)(smem + LDS_W2_OFF);
  for (int i = tid; i < DDIM * EDIM; i += 256) sW2[i] = W2[i];
#endif
}

extern "C" __global__ void __launch_bounds__(256)
moe_router_kernel(const float* __restrict__ x,
                  const float* __restrict__ W1,
                  const float* __restrict__ b1,
                  const float* __restrict__ W2,
                  const float* __restrict__ b2,
                  float* __restrict__ gates,
                  int* __restrict__ topidx)
{
  extern __shared__ char smem[];
  const int tid   = threadIdx.x;
  const int lane  = tid & 31;
  const int wave  = tid >> 5;
  const int lmod  = lane & 15;   // column / row-within-16
  const int lhalf = lane >> 4;   // which K-half this lane holds

  // ---- stage W2 via async-to-LDS (issue first, overlap with W1 staging) ----
  stage_w2(W2, smem, tid);

  // ---- stage W1^T (bf16) into LDS: thread t owns LDS row n=t (column n of W1) ----
  {
    __bf16* dst = (__bf16*)(smem + tid * LDS_W1_STRIDE);
#pragma unroll 8
    for (int k = 0; k < DDIM; k += 2) {
      float f0 = W1[(k + 0) * HDIM + tid];   // coalesced across threads per k
      float f1 = W1[(k + 1) * HDIM + tid];
      union { __bf16 h[2]; unsigned int u; } p;
      p.h[0] = (__bf16)f0;
      p.h[1] = (__bf16)f1;
      *(unsigned int*)(dst + k) = p.u;       // ds_store_b32
    }
  }

#if defined(HAVE_ASYNC_LDS)
#if defined(HAVE_WAIT_ASYNC)
  __builtin_amdgcn_s_wait_asynccnt(0);
#else
  asm volatile("s_wait_asynccnt 0" ::: "memory");
#endif
#endif
  __syncthreads();

  // ---- GEMM1: h = x @ W1, 16 rows per wave, f32 accum via bf16 WMMA ----
  const long rowbase = (long)blockIdx.x * 128 + (long)wave * 16;
  const float* xrow  = x + (rowbase + lmod) * DDIM;   // this lane's A row (M = lmod)
  const int kbase    = lhalf * 8;                     // A layout: half-wave K offset

  v8f acc[16];
  const v8f vzero = {0.f, 0.f, 0.f, 0.f, 0.f, 0.f, 0.f, 0.f};
#pragma unroll
  for (int t = 0; t < 16; ++t) acc[t] = vzero;

  for (int ks = 0; ks < 8; ++ks) {                    // K = 8 * 32 = 256
    // A fragment: elems 0..7 = K[kbase..kbase+8), elems 8..15 = K[kbase+16..+24)
    const float* ap = xrow + ks * 32 + kbase;
    v8f f01 = *(const v8f*)(ap);        // 32B, 32B-aligned
    v8f f23 = *(const v8f*)(ap + 16);
    v8bf alo = __builtin_convertvector(f01, v8bf);
    v8bf ahi = __builtin_convertvector(f23, v8bf);
    v16bf a = __builtin_shufflevector(alo, ahi,
                                      0, 1, 2, 3, 4, 5, 6, 7,
                                      8, 9, 10, 11, 12, 13, 14, 15);
#pragma unroll
    for (int t = 0; t < 16; ++t) {
      // B fragment: column n = t*16 + lmod, elems 0..7 = K[ks*32+lhalf*8..+8),
      // elems 8..15 = K[ks*32+16+lhalf*8..+8)  (same K order as A)
      const char* bp = smem + (t * 16 + lmod) * LDS_W1_STRIDE + (ks * 32 + lhalf * 8) * 2;
      union { u32x4 q[2]; v16bf bf; } bu;
      bu.q[0] = *(const u32x4*)(bp);        // ds_load_b128
      bu.q[1] = *(const u32x4*)(bp + 32);   // ds_load_b128
      acc[t] = __builtin_amdgcn_wmma_f32_16x16x32_bf16(
          false, a, false, bu.bf, (short)0, acc[t], false, false);
    }
  }

  // ---- GEMM2: logits = relu(acc + b1) @ W2 (per-lane partials over 16 cols) ----
  const float* sW2 = (const float*)(smem + LDS_W2_OFF);
  float part[8][5];
#pragma unroll
  for (int r = 0; r < 8; ++r)
#pragma unroll
    for (int e = 0; e < 5; ++e) part[r][e] = 0.f;

#pragma unroll
  for (int t = 0; t < 16; ++t) {
    const int col = t * 16 + lmod;
    const float bb = b1[col];
    float w2v[5];
#pragma unroll
    for (int e = 0; e < 5; ++e) w2v[e] = sW2[col * EDIM + e];
#pragma unroll
    for (int r = 0; r < 8; ++r) {
      float hv = acc[t][r] + bb;
      hv = fmaxf(hv, 0.f);
#pragma unroll
      for (int e = 0; e < 5; ++e) part[r][e] = fmaf(hv, w2v[e], part[r][e]);
    }
  }

  // reduce partials across the 16 lanes of each half-wave
#pragma unroll
  for (int r = 0; r < 8; ++r)
#pragma unroll
    for (int e = 0; e < 5; ++e) {
      float v = part[r][e];
      v += __shfl_xor(v, 1, 32);
      v += __shfl_xor(v, 2, 32);
      v += __shfl_xor(v, 4, 32);
      v += __shfl_xor(v, 8, 32);
      part[r][e] = v;
    }

  // ---- top-2 + softmax + scatter (lanes 0 and 16 handle 8 rows each) ----
  if (lmod == 0) {
    const long row0 = rowbase + lhalf * 8;
    float b2v[5];
#pragma unroll
    for (int e = 0; e < 5; ++e) b2v[e] = b2[e];
#pragma unroll
    for (int r = 0; r < 8; ++r) {
      float l[5];
#pragma unroll
      for (int e = 0; e < 5; ++e) l[e] = part[r][e] + b2v[e];
      int i0 = 0; float v0 = l[0];
      int i1 = -1; float v1 = -__builtin_inff();
#pragma unroll
      for (int e = 1; e < 5; ++e) {
        if (l[e] > v0)      { v1 = v0; i1 = i0; v0 = l[e]; i0 = e; }
        else if (l[e] > v1) { v1 = l[e]; i1 = e; }
      }
      float ex = __expf(v1 - v0);      // v0 >= v1, stable 2-way softmax
      float s  = 1.f + ex;
      float g0 = 1.f / s;
      float g1 = ex / s;
      const long row = row0 + r;
#pragma unroll
      for (int e = 0; e < 5; ++e) {
        float g = (e == i0) ? g0 : ((e == i1) ? g1 : 0.f);
        gates[row * EDIM + e] = g;
      }
      topidx[row * 2 + 0] = i0;
      topidx[row * 2 + 1] = i1;
    }
  }
}

extern "C" void kernel_launch(void* const* d_in, const int* in_sizes, int n_in,
                              void* d_out, int out_size, void* d_ws, size_t ws_size,
                              hipStream_t stream)
{
  (void)in_sizes; (void)n_in; (void)out_size; (void)d_ws; (void)ws_size;
  const float* x  = (const float*)d_in[0];
  const float* W1 = (const float*)d_in[1];
  const float* b1 = (const float*)d_in[2];
  const float* W2 = (const float*)d_in[3];
  const float* b2 = (const float*)d_in[4];

  float* gates  = (float*)d_out;                              // [B,5] f32
  int*   topidx = (int*)(gates + (size_t)B_ROWS * EDIM);      // [B,2] i32

  dim3 grid(B_ROWS / 128);   // 128 rows per workgroup (8 waves x 16 rows)
  dim3 block(256);
  moe_router_kernel<<<grid, block, LDS_BYTES, stream>>>(x, W1, b1, W2, b2, gates, topidx);
}
